// CoAttention_54013508714705
// MI455X (gfx1250) — compile-verified, matrix operand order checked by reading, not run on hardware
//
#include <hip/hip_runtime.h>
#include <math.h>

typedef _Float16 f16;
typedef __attribute__((ext_vector_type(16))) _Float16 v16h;
typedef __attribute__((ext_vector_type(8)))  float    v8f;
typedef int v4i_ __attribute__((vector_size(16)));

#define AS1 __attribute__((address_space(1)))
#define AS3 __attribute__((address_space(3)))

#define DIM   192
#define HEADS 8
#define CH    24           // channels per head
#define NHW   16384        // H*W
#define HH    128
#define WW    128

#if defined(__has_builtin)
#if __has_builtin(__builtin_amdgcn_global_load_async_to_lds_b128) && \
    __has_builtin(__builtin_amdgcn_s_wait_asynccnt)
#define HAVE_ASYNC_LDS 1
#endif
#endif
#ifndef HAVE_ASYNC_LDS
#define HAVE_ASYNC_LDS 0
#endif

// ---------------------------------------------------------------------------
// Transpose + convert: in fp32 [DIM][NHW] -> out f16 [NHW][DIM]
// ---------------------------------------------------------------------------
__global__ __launch_bounds__(256)
void transcvt_kernel(const float* __restrict__ in, f16* __restrict__ out) {
    __shared__ f16 tile[32][33];
    const int bx = blockIdx.x * 32;          // n base
    const int by = blockIdx.y * 32;          // c base
    const int tx = threadIdx.x & 31;
    const int ty = threadIdx.x >> 5;         // 8 rows per pass
    #pragma unroll
    for (int i = 0; i < 32; i += 8)
        tile[ty + i][tx] = (f16)in[(size_t)(by + ty + i) * NHW + bx + tx];
    __syncthreads();
    #pragma unroll
    for (int i = 0; i < 32; i += 8)
        out[(size_t)(bx + ty + i) * DIM + by + tx] = tile[tx][ty + i];
}

// ---------------------------------------------------------------------------
// X tile staging: 128 tokens x 32 k-slice of Xt[n][k] -> LDS, 16B chunks.
// Async-to-LDS (CDNA5 path) when available; register copy fallback.
// ---------------------------------------------------------------------------
__device__ inline void stage_x(f16 (*__restrict__ dst)[32],
                               const f16* __restrict__ Xt,
                               int n0, int k0, int tid) {
    #pragma unroll
    for (int i = 0; i < 2; ++i) {
        const int chunk = tid + i * 256;     // 0..511 (512 x 16B = 8KB)
        const int r = chunk >> 2;            // 4 chunks per 64B row
        const int c = (chunk & 3) * 8;
        const f16* g = Xt + (size_t)(n0 + r) * DIM + k0 + c;
        f16* l = &dst[r][c];
#if HAVE_ASYNC_LDS
        __builtin_amdgcn_global_load_async_to_lds_b128(
            (AS1 v4i_*)g, (AS3 v4i_*)l, 0, 0);
#else
        *(float4*)l = *(const float4*)g;
#endif
    }
}

__device__ inline void wait_stage_x() {
#if HAVE_ASYNC_LDS
    __builtin_amdgcn_s_wait_asynccnt(0);
#endif
}

// ---------------------------------------------------------------------------
// Tiled WMMA GEMM: Out[o, n] = sum_k W[o, k] * Xt[n, k]
//   Xt is token-major f16 (K contiguous). Block tile 64 o x 128 n.
//   W panel staged once; X double-buffered via async LDS DMA.
//   OUT_T=false -> Out[o*NHW+n] ; OUT_T=true -> Out[n*DIM+o].
// ---------------------------------------------------------------------------
template <typename TW, typename TO, bool OUT_T>
__global__ __launch_bounds__(256)
void gemm_wmma_kernel(const TW* __restrict__ W, const f16* __restrict__ Xt,
                      TO* __restrict__ Out) {
    __shared__ __align__(16) f16 sW[64][DIM];      // 24 KB, full K panel
    __shared__ __align__(16) f16 sX[2][128][32];   // 2 x 8 KB, [n][k]

    const int tid  = threadIdx.x;
    const int wave = tid >> 5;
    const int lane = tid & 31;
    const int n0   = blockIdx.x * 128;
    const int o0   = blockIdx.y * 64;
    const int wo   = wave >> 1;        // o sub-tile (16 rows)
    const int wn   = (wave & 1) * 64;  // n half
    const int m    = lane & 15;
    const int hi   = lane >> 4;

    // stage whole W panel (fp32 or f16 -> f16)
    for (int i = tid; i < 64 * DIM; i += 256) {
        const int r = i / DIM, c = i - r * DIM;
        sW[r][c] = (f16)(float)W[(o0 + r) * DIM + c];
    }
    stage_x(sX[0], Xt, n0, 0, tid);

    v8f acc[4] = {};

    #pragma unroll
    for (int s = 0; s < 6; ++s) {          // K steps of 32 (DIM = 192)
        wait_stage_x();
        __syncthreads();                   // buffer s&1 ready; s+1 buffer free
        if (s < 5) stage_x(sX[(s + 1) & 1], Xt, n0, (s + 1) * 32, tid);

        // A fragment: lane m = o row; element e<8 -> K=hi*8+e, e>=8 -> 16+hi*8+(e-8)
        v16h a;
        #pragma unroll
        for (int j = 0; j < 8; ++j) {
            a[j]     = sW[wo * 16 + m][s * 32 + hi * 8 + j];
            a[j + 8] = sW[wo * 16 + m][s * 32 + 16 + hi * 8 + j];
        }
        #pragma unroll
        for (int t = 0; t < 4; ++t) {
            const f16* bp = &sX[s & 1][wn + t * 16 + m][0];
            v16h b;
            #pragma unroll
            for (int j = 0; j < 8; ++j) {
                b[j]     = bp[hi * 8 + j];
                b[j + 8] = bp[16 + hi * 8 + j];
            }
            acc[t] = __builtin_amdgcn_wmma_f32_16x16x32_f16(
                false, a, false, b, (short)0, acc[t], false, false);
        }
    }

    // C/D layout: VGPR r, lanes 0-15 -> M=r, lanes 16-31 -> M=8+r; N = lane&15
    #pragma unroll
    for (int t = 0; t < 4; ++t) {
        #pragma unroll
        for (int r = 0; r < 8; ++r) {
            const int row = o0 + wo * 16 + hi * 8 + r;
            const int col = n0 + wn + t * 16 + m;
            if (OUT_T) Out[(size_t)col * DIM + row] = (TO)acc[t][r];
            else       Out[(size_t)row * NHW + col] = (TO)acc[t][r];
        }
    }
}

// ---------------------------------------------------------------------------
// Depthwise 3x3, zero padding — channel-major layout [c][h][w]
// ---------------------------------------------------------------------------
__global__ __launch_bounds__(256)
void dwconv_cn_kernel(const f16* __restrict__ in, const float* __restrict__ w,
                      f16* __restrict__ out) {
    const int idx = blockIdx.x * 256 + threadIdx.x;  // DIM*NHW
    const int x = idx & (WW - 1);
    const int y = (idx >> 7) & (HH - 1);
    const int c = idx >> 14;
    const float* wc = w + c * 9;
    const f16* pc = in + (size_t)c * NHW;
    float s = 0.f;
    #pragma unroll
    for (int dy = -1; dy <= 1; ++dy) {
        const int yy = y + dy;
        if (yy < 0 || yy >= HH) continue;
        #pragma unroll
        for (int dx = -1; dx <= 1; ++dx) {
            const int xx = x + dx;
            if (xx < 0 || xx >= WW) continue;
            s += wc[(dy + 1) * 3 + (dx + 1)] * (float)pc[yy * WW + xx];
        }
    }
    out[idx] = (f16)s;
}

// Depthwise 3x3 — token-major layout [h][w][c] (c fastest, fully coalesced)
__global__ __launch_bounds__(256)
void dwconv_nc_kernel(const f16* __restrict__ in, const float* __restrict__ w,
                      f16* __restrict__ out) {
    const int idx = blockIdx.x * 256 + threadIdx.x;  // DIM*NHW, c fastest
    const int n = idx / DIM;
    const int c = idx - n * DIM;
    const int x = n & (WW - 1);
    const int y = n >> 7;
    const float* wc = w + c * 9;
    float s = 0.f;
    #pragma unroll
    for (int dy = -1; dy <= 1; ++dy) {
        const int yy = y + dy;
        if (yy < 0 || yy >= HH) continue;
        #pragma unroll
        for (int dx = -1; dx <= 1; ++dx) {
            const int xx = x + dx;
            if (xx < 0 || xx >= WW) continue;
            s += wc[(dy + 1) * 3 + (dx + 1)] * (float)in[(size_t)(yy * WW + xx) * DIM + c];
        }
    }
    out[idx] = (f16)s;
}

// ---------------------------------------------------------------------------
// Inverse L2 norms over tokens per channel; Q (bx<192) and K ([c][n] layout).
// ---------------------------------------------------------------------------
__global__ __launch_bounds__(256)
void norm_kernel(const f16* __restrict__ Q, const f16* __restrict__ Kk,
                 float* __restrict__ inv) {
    const int bx = blockIdx.x;  // 0..383
    const f16* src = (bx < DIM ? Q : Kk) + (size_t)(bx % DIM) * NHW;
    float s = 0.f;
    for (int i = threadIdx.x; i < NHW; i += 256) {
        float v = (float)src[i];
        s += v * v;
    }
    __shared__ float red[256];
    red[threadIdx.x] = s;
    __syncthreads();
    for (int off = 128; off > 0; off >>= 1) {
        if (threadIdx.x < off) red[threadIdx.x] += red[threadIdx.x + off];
        __syncthreads();
    }
    if (threadIdx.x == 0) inv[bx] = 1.0f / fmaxf(sqrtf(red[0]), 1e-12f);
}

// ---------------------------------------------------------------------------
// Per-head attention: 24x24 (padded 32x32) WMMA GEMM over 16384 tokens,
// cross-wave LDS float atomics, softmax with norms/temperature folded in.
// ---------------------------------------------------------------------------
__device__ inline v16h frag_row(const f16* __restrict__ rowp, int n0, int hi) {
    v16h r;
    #pragma unroll
    for (int j = 0; j < 8; ++j) {
        r[j]     = rowp[n0 + hi * 8 + j];
        r[j + 8] = rowp[n0 + 16 + hi * 8 + j];
    }
    return r;
}

__global__ __launch_bounds__(256)
void attn_kernel(const f16* __restrict__ Q, const f16* __restrict__ Kk,
                 const float* __restrict__ inv, const float* __restrict__ temp,
                 float* __restrict__ attn_out) {
    const int h = blockIdx.x;
    const f16* q = Q + (size_t)h * CH * NHW;
    const f16* k = Kk + (size_t)h * CH * NHW;

    __shared__ float S[32][32];
    for (int i = threadIdx.x; i < 1024; i += 256) ((float*)S)[i] = 0.f;
    __syncthreads();

    const int wave = threadIdx.x >> 5;
    const int lane = threadIdx.x & 31;
    const int m  = lane & 15;
    const int hi = lane >> 4;

    v8f acc[2][2] = {};
    const int nbeg = wave * (NHW / 8);
    const int nend = nbeg + (NHW / 8);
    for (int n0 = nbeg; n0 < nend; n0 += 32) {
        v16h afr[2], bfr[2];
        v16h zero = {};
        #pragma unroll
        for (int t = 0; t < 2; ++t) {
            const int row = t * 16 + m;   // pad rows >= 24 with zeros
            afr[t] = (row < CH) ? frag_row(q + (size_t)row * NHW, n0, hi) : zero;
            bfr[t] = (row < CH) ? frag_row(k + (size_t)row * NHW, n0, hi) : zero;
        }
        #pragma unroll
        for (int ti = 0; ti < 2; ++ti)
            #pragma unroll
            for (int tj = 0; tj < 2; ++tj)
                acc[ti][tj] = __builtin_amdgcn_wmma_f32_16x16x32_f16(
                    false, afr[ti], false, bfr[tj], (short)0, acc[ti][tj],
                    false, false);
    }
    #pragma unroll
    for (int ti = 0; ti < 2; ++ti)
        #pragma unroll
        for (int tj = 0; tj < 2; ++tj)
            #pragma unroll
            for (int r = 0; r < 8; ++r)
                atomicAdd(&S[ti * 16 + hi * 8 + r][tj * 16 + m], acc[ti][tj][r]);
    __syncthreads();

    if (threadIdx.x < CH) {
        const int c = threadIdx.x;
        const float th = temp[h];
        const float qi = inv[h * CH + c];
        float vals[CH];
        float mx = -3.0e38f;
        #pragma unroll
        for (int d = 0; d < CH; ++d) {
            float v = S[c][d] * qi * inv[DIM + h * CH + d] * th;
            vals[d] = v;
            mx = fmaxf(mx, v);
        }
        float sum = 0.f;
        #pragma unroll
        for (int d = 0; d < CH; ++d) {
            vals[d] = expf(vals[d] - mx);
            sum += vals[d];
        }
        const float rs = 1.0f / sum;
        #pragma unroll
        for (int d = 0; d < CH; ++d)
            attn_out[h * CH * CH + c * CH + d] = vals[d] * rs;
    }
}

// ---------------------------------------------------------------------------
// M[o][d] = sum_j proj[o][h*24+j] * attn[h][j][dl]  (h = d/24)
// so that final output = M @ V in one GEMM.
// ---------------------------------------------------------------------------
__global__ __launch_bounds__(256)
void foldm_kernel(const float* __restrict__ proj, const float* __restrict__ attn,
                  f16* __restrict__ M) {
    const int idx = blockIdx.x * 256 + threadIdx.x;  // DIM*DIM
    const int d = idx % DIM;
    const int o = idx / DIM;
    const int h = d / CH;
    const int dl = d % CH;
    float s = 0.f;
    #pragma unroll
    for (int j = 0; j < CH; ++j)
        s += proj[o * DIM + h * CH + j] * attn[h * CH * CH + j * CH + dl];
    M[o * DIM + d] = (f16)s;
}

// ---------------------------------------------------------------------------
extern "C" void kernel_launch(void* const* d_in, const int* in_sizes, int n_in,
                              void* d_out, int out_size, void* d_ws, size_t ws_size,
                              hipStream_t stream) {
    (void)in_sizes; (void)n_in; (void)out_size; (void)ws_size;
    const float* x_feat      = (const float*)d_in[0];
    const float* y_feat      = (const float*)d_in[1];
    const float* q_pw        = (const float*)d_in[2];
    const float* q_dw        = (const float*)d_in[3];
    const float* k_pw        = (const float*)d_in[4];
    const float* k_dw        = (const float*)d_in[5];
    const float* v_pw        = (const float*)d_in[6];
    const float* v_dw        = (const float*)d_in[7];
    const float* proj_w      = (const float*)d_in[8];
    const float* temperature = (const float*)d_in[9];
    float* out = (float*)d_out;

    const size_t PLANE = (size_t)DIM * NHW;
    char* ws = (char*)d_ws;
    f16*   bufXt = (f16*)ws;                         // features, f16 [n][c]
    f16*   bufPW = (f16*)(ws + PLANE * 2);           // pointwise result
    f16*   bufQ  = (f16*)(ws + PLANE * 4);           // [c][n]
    f16*   bufK  = (f16*)(ws + PLANE * 6);           // [c][n]
    f16*   bufV  = (f16*)(ws + PLANE * 8);           // [n][c]
    float* invn  = (float*)(ws + PLANE * 10);        // 384 floats
    float* attnb = invn + 512;                       // 8*24*24 floats
    f16*   Mbuf  = (f16*)(attnb + HEADS * CH * CH);  // 192*192 f16

    const dim3 tgrid(NHW / 32, DIM / 32);
    const dim3 ggrid(NHW / 128, DIM / 64);
    const int dwb = (DIM * NHW) / 256;

    for (int b = 0; b < 8; ++b) {
        const float* xb = x_feat + (size_t)b * PLANE;
        const float* yb = y_feat + (size_t)b * PLANE;

        transcvt_kernel<<<tgrid, 256, 0, stream>>>(xb, bufXt);
        gemm_wmma_kernel<float, f16, false><<<ggrid, 256, 0, stream>>>(q_pw, bufXt, bufPW);
        dwconv_cn_kernel<<<dwb, 256, 0, stream>>>(bufPW, q_dw, bufQ);

        transcvt_kernel<<<tgrid, 256, 0, stream>>>(yb, bufXt);
        gemm_wmma_kernel<float, f16, false><<<ggrid, 256, 0, stream>>>(k_pw, bufXt, bufPW);
        dwconv_cn_kernel<<<dwb, 256, 0, stream>>>(bufPW, k_dw, bufK);

        gemm_wmma_kernel<float, f16, true><<<ggrid, 256, 0, stream>>>(v_pw, bufXt, bufPW);
        dwconv_nc_kernel<<<dwb, 256, 0, stream>>>(bufPW, v_dw, bufV);

        norm_kernel<<<2 * DIM, 256, 0, stream>>>(bufQ, bufK, invn);
        attn_kernel<<<HEADS, 256, 0, stream>>>(bufQ, bufK, invn, temperature, attnb);
        foldm_kernel<<<(DIM * DIM) / 256, 256, 0, stream>>>(proj_w, attnb, Mbuf);

        // fused (attn @ v) + projection, fp32 straight to d_out
        gemm_wmma_kernel<f16, float, false><<<ggrid, 256, 0, stream>>>(
            Mbuf, bufV, out + (size_t)b * PLANE);
    }
}
